// HierarchicalMoEMLPBlock_558345749171
// MI455X (gfx1250) — matrix operand order repaired and checked by reading, not compiled, not run on hardware
//
#include <hip/hip_runtime.h>
#include <hip/hip_bf16.h>

#define NTOK 1024
#define DDIM 768
#define MDIM 3072
#define GN   4
#define EN   8
#define GE   (GN*EN)
#define TILE_N 32
#define CHUNK  128
#define XPAD   776   // LDS row stride (bf16) for x tile; 1552 B rows: 16B-aligned, bank-friendly
#define HPAD   136   // LDS row stride (bf16) for h chunk; 272 B rows: 16B-aligned, bank-friendly
#define KB1    (DDIM/32)   // 24  k-tiles for GEMM1
#define NB1    (MDIM/16)   // 192 n-tiles for GEMM1
#define KB2    (MDIM/32)   // 96  k-tiles for GEMM2
#define NB2    (DDIM/16)   // 48  n-tiles for GEMM2

typedef __attribute__((ext_vector_type(16))) __bf16 v16bf;
typedef __attribute__((ext_vector_type(8)))  float  v8f;

union alignas(32) BFrag {
    unsigned int u[8];
    uint4        q[2];
    v16bf        v;
};

__device__ __forceinline__ unsigned short f2bf(float f) {
    unsigned int u = __float_as_uint(f);
    unsigned int r = u + 0x7FFFu + ((u >> 16) & 1u);   // round-to-nearest-even
    return (unsigned short)(r >> 16);
}
__device__ __forceinline__ unsigned int pack2(float lo, float hi) {
    return ((unsigned int)f2bf(hi) << 16) | (unsigned int)f2bf(lo);
}

// branchless gelu: Abramowitz-Stegun 7.1.26 erf (|err| < 1.5e-7)
__device__ __forceinline__ float gelu_f(float t) {
    float ax = fabsf(t);
    float u  = 1.0f / fmaf(0.3275911f, ax, 1.0f);
    float p  = fmaf(u, 1.061405429f, -1.453152027f);
    p = fmaf(u, p, 1.421413741f);
    p = fmaf(u, p, -0.284496736f);
    p = fmaf(u, p, 0.254829592f);
    p = p * u;
    float er = fmaf(-p, __expf(-ax * ax), 1.0f);
    er = copysignf(er, t);
    return 0.5f * t * (1.0f + er);
}

// ---------------------------------------------------------------- zero y
__global__ void zero_kernel(float* __restrict__ y, int n) {
    int i = blockIdx.x * blockDim.x + threadIdx.x;
    if (i < n) y[i] = 0.0f;
}

// ---------------------------------------------------------------- x -> bf16
__global__ void cvt_kernel(const float* __restrict__ x, unsigned short* __restrict__ xbf, int n) {
    int i = blockIdx.x * blockDim.x + threadIdx.x;
    if (i < n) xbf[i] = f2bf(x[i]);
}

// ------------------------------------------------- weight pre-pack (fragment-major bf16)
// pw1: [ge][nb<NB1][kb<KB1][dword<8][lane<32]; lane dword v = bf16{k,k+1} with
//   k = kb*32 + (lane>>4)*8 + (v>>2)*16 + (v&3)*2, col = nb*16 + (lane&15)   (WMMA B-fragment order)
__global__ void pack_w1_kernel(const float* __restrict__ W1, unsigned int* __restrict__ pw1) {
    int i = blockIdx.x * blockDim.x + threadIdx.x;           // total GE*NB1*KB1*256
    int lane = i & 31;
    int d    = (i >> 5) & 7;
    int rest = i >> 8;
    int kb = rest % KB1; rest /= KB1;
    int nb = rest % NB1;
    int ge = rest / NB1;
    int k   = kb * 32 + (lane >> 4) * 8 + (d >> 2) * 16 + (d & 3) * 2;
    int col = nb * 16 + (lane & 15);
    const float* p = W1 + ((size_t)ge * DDIM + k) * MDIM + col;
    pw1[i] = pack2(p[0], p[MDIM]);
}
// pw2: [ge][nb<NB2][kb<KB2][dword][lane]
__global__ void pack_w2_kernel(const float* __restrict__ W2, unsigned int* __restrict__ pw2) {
    int i = blockIdx.x * blockDim.x + threadIdx.x;           // total GE*NB2*KB2*256
    int lane = i & 31;
    int d    = (i >> 5) & 7;
    int rest = i >> 8;
    int kb = rest % KB2; rest /= KB2;
    int nb = rest % NB2;
    int ge = rest / NB2;
    int k   = kb * 32 + (lane >> 4) * 8 + (d >> 2) * 16 + (d & 3) * 2;
    int col = nb * 16 + (lane & 15);
    const float* p = W2 + ((size_t)ge * MDIM + k) * DDIM + col;
    pw2[i] = pack2(p[0], p[DDIM]);
}

// ---------------------------------------------------------------- gating
__global__ __launch_bounds__(128)
void gate_kernel(const float* __restrict__ x,
                 const float* __restrict__ Wg, const float* __restrict__ bg,
                 const float* __restrict__ We, const float* __restrict__ be,
                 float* __restrict__ combine) {
    __shared__ float xs[DDIM];
    __shared__ float logits[GN + GN * EN];   // 36
    const int n = blockIdx.x;
    for (int i = threadIdx.x; i < DDIM; i += blockDim.x) xs[i] = x[(size_t)n * DDIM + i];
    __syncthreads();
    const int t = threadIdx.x;
    if (t < GN) {
        float s = bg[t];
        for (int d = 0; d < DDIM; ++d) s += xs[d] * Wg[d * GN + t];
        logits[t] = s;
    } else if (t < GN + GN * EN) {
        int idx = t - GN;
        int g = idx >> 3, e = idx & 7;
        float s = be[g * EN + e];
        const float* w = We + (size_t)g * DDIM * EN + e;
        for (int d = 0; d < DDIM; ++d) s += xs[d] * w[(size_t)d * EN];
        logits[t] = s;
    }
    __syncthreads();
    if (t == 0) {
        float gw[GN];
        for (int g = 0; g < GN; ++g) gw[g] = 0.0f;
        int i0 = 0; float v0 = logits[0];
        for (int g = 1; g < GN; ++g) if (logits[g] > v0) { v0 = logits[g]; i0 = g; }
        int i1 = -1; float v1 = -1e30f;
        for (int g = 0; g < GN; ++g) { if (g == i0) continue; if (logits[g] > v1) { v1 = logits[g]; i1 = g; } }
        float e1 = expf(v1 - v0);
        float s2 = 1.0f + e1;
        float w0 = 1.0f / s2, w1 = e1 / s2;
        gw[i0] = (w0 >= 1e-6f) ? w0 : 0.0f;
        gw[i1] = (w1 >= 1e-6f) ? w1 : 0.0f;
        for (int g = 0; g < GN; ++g) {
            const float* el = logits + GN + g * EN;
            int sel[4]; float sv[4];
            unsigned used = 0;
            for (int k = 0; k < 4; ++k) {
                int bi = -1; float bv = -1e30f;
                for (int e = 0; e < EN; ++e)
                    if (!((used >> e) & 1u) && el[e] > bv) { bv = el[e]; bi = e; }
                sel[k] = bi; sv[k] = bv; used |= 1u << bi;
            }
            float ssum = 0.0f, ex[4];
            for (int k = 0; k < 4; ++k) { ex[k] = expf(sv[k] - sv[0]); ssum += ex[k]; }
            float ew[EN];
            for (int e = 0; e < EN; ++e) ew[e] = 0.0f;
            for (int k = 0; k < 4; ++k) {
                float w = ex[k] / ssum;
                ew[sel[k]] = (w >= 1e-6f) ? w : 0.0f;
            }
            for (int e = 0; e < EN; ++e)
                combine[(size_t)n * GE + g * EN + e] = gw[g] * ew[e];
        }
    }
}

// ---------------------------------------------------------------- fused expert MLP
// grid: (N/TILE_N, G*E) ; block: 256 threads (8 wave32)
template <bool PACKED>
__global__ __launch_bounds__(256)
void moe_ffn_kernel(const unsigned short* __restrict__ xbf,
                    const float* __restrict__ W1, const float* __restrict__ b1,
                    const float* __restrict__ W2, const float* __restrict__ b2,
                    const unsigned int* __restrict__ pw1,
                    const unsigned int* __restrict__ pw2,
                    const float* __restrict__ combine,
                    float* __restrict__ y) {
    __shared__ unsigned short xs[TILE_N * XPAD];   // 32x768 bf16 x-tile
    __shared__ unsigned short hs[TILE_N * HPAD];   // 32x128 bf16 h-chunk
    __shared__ float cw_s[TILE_N];

    const int tile = blockIdx.x;
    const int ge   = blockIdx.y;           // expert id 0..31
    const int tid  = threadIdx.x;
    const int wave = tid >> 5;
    const int lane = tid & 31;
    const int n16  = lane & 15;
    const int half = lane >> 4;
    const int row0 = tile * TILE_N;

    // stage x tile (bf16, 2 elems per b32)
    for (int i = tid; i < TILE_N * (DDIM / 2); i += 256) {
        int r = i / (DDIM / 2);
        int c = (i % (DDIM / 2)) * 2;
        unsigned int v = *(const unsigned int*)(xbf + (size_t)(row0 + r) * DDIM + c);
        *(unsigned int*)(xs + r * XPAD + c) = v;
    }
    if (tid < TILE_N) cw_s[tid] = combine[(size_t)(row0 + tid) * GE + ge];
    __syncthreads();

    const float* W1e = W1 + (size_t)ge * DDIM * MDIM;
    const float* W2e = W2 + (size_t)ge * MDIM * DDIM;
    const float* b1e = b1 + (size_t)ge * MDIM;
    const float* b2e = b2 + (size_t)ge * DDIM;
    const unsigned int* pw1e = pw1 + (size_t)ge * NB1 * KB1 * 256;
    const unsigned int* pw2e = pw2 + (size_t)ge * NB2 * KB2 * 256;

    // persistent second-GEMM accumulators: 2 row-tiles x 6 col-tiles per wave
    v8f acc[2][6];
    for (int r = 0; r < 2; ++r)
        for (int j = 0; j < 6; ++j)
            acc[r][j] = v8f{};

    // A-fragment LDS bases (elements, 16B-aligned) for the two 16-row tiles
    const int abase0 = n16 * XPAD + half * 8;
    const int abase1 = (16 + n16) * XPAD + half * 8;

    for (int mc = 0; mc < MDIM / CHUNK; ++mc) {
        // ---------------- GEMM 1: h = gelu(x @ W1 + b1), 128 m-cols per chunk --
        // each wave: one 16-col slice (ct = wave), BOTH 16-row tiles -> B reused 2x
        v8f hacc[2];
        hacc[0] = v8f{};
        hacc[1] = v8f{};
        const int nb1  = mc * 8 + wave;                 // B col-tile index
        const int mcol = nb1 * 16 + n16;                // this lane's W1 column
        const unsigned int* pb1 = pw1e + (size_t)nb1 * KB1 * 256 + lane * 8;
        // prefetch next chunk's first W1 tile (sequential packed stream)
        __builtin_prefetch(pw1e + (size_t)(nb1 + 8) * KB1 * 256 + lane * 8, 0, 3);

        BFrag a0[2], a1[2], b[2];
        {   // preload k-step 0
            const unsigned short* ap0 = xs + abase0;
            const unsigned short* ap1 = xs + abase1;
            a0[0].q[0] = *(const uint4*)(ap0);
            a0[0].q[1] = *(const uint4*)(ap0 + 16);
            a1[0].q[0] = *(const uint4*)(ap1);
            a1[0].q[1] = *(const uint4*)(ap1 + 16);
            if (PACKED) {
                b[0].q[0] = *(const uint4*)(pb1);
                b[0].q[1] = *(const uint4*)(pb1 + 4);
            } else {
#pragma unroll
                for (int v = 0; v < 8; ++v) {
                    int kk = half * 8 + (v >> 2) * 16 + (v & 3) * 2;
                    const float* p = W1e + (size_t)kk * MDIM + mcol;
                    b[0].u[v] = pack2(p[0], p[MDIM]);
                }
            }
        }
#pragma unroll
        for (int ks = 0; ks < KB1; ++ks) {
            const int cur = ks & 1, nxt = cur ^ 1;
            if (ks + 1 < KB1) {   // double-buffer: fetch next k-step while computing
                const unsigned short* ap0 = xs + abase0 + (ks + 1) * 32;
                const unsigned short* ap1 = xs + abase1 + (ks + 1) * 32;
                a0[nxt].q[0] = *(const uint4*)(ap0);
                a0[nxt].q[1] = *(const uint4*)(ap0 + 16);
                a1[nxt].q[0] = *(const uint4*)(ap1);
                a1[nxt].q[1] = *(const uint4*)(ap1 + 16);
                if (PACKED) {
                    const unsigned int* pn = pb1 + (ks + 1) * 256;
                    b[nxt].q[0] = *(const uint4*)(pn);
                    b[nxt].q[1] = *(const uint4*)(pn + 4);
                } else {
#pragma unroll
                    for (int v = 0; v < 8; ++v) {
                        int kk = (ks + 1) * 32 + half * 8 + (v >> 2) * 16 + (v & 3) * 2;
                        const float* p = W1e + (size_t)kk * MDIM + mcol;
                        b[nxt].u[v] = pack2(p[0], p[MDIM]);
                    }
                }
            }
            hacc[0] = __builtin_amdgcn_wmma_f32_16x16x32_bf16(
                false, a0[cur].v, false, b[cur].v, (short)0, hacc[0], false, false);
            hacc[1] = __builtin_amdgcn_wmma_f32_16x16x32_bf16(
                false, a1[cur].v, false, b[cur].v, (short)0, hacc[1], false, false);
        }
        {
            const float bias = b1e[mcol];
#pragma unroll
            for (int rt = 0; rt < 2; ++rt) {
#pragma unroll
                for (int v = 0; v < 8; ++v) {
                    float t = gelu_f(hacc[rt][v] + bias);
                    int r = rt * 16 + half * 8 + v;
                    int c = wave * 16 + n16;
                    hs[r * HPAD + c] = f2bf(t);
                }
            }
        }
        __syncthreads();

        // ---------------- GEMM 2: acc += h_chunk @ W2_chunk --------------------
        // prefetch next chunk's W2 stream for this wave's first col tile
        __builtin_prefetch(pw2e + ((size_t)(wave * 6) * KB2 + (mc + 1) * 4) * 256 + lane * 8, 0, 3);
        for (int ks = 0; ks < CHUNK / 32; ++ks) {
            BFrag ha0, ha1;
            {
                const unsigned short* hp0 = hs + n16 * HPAD + ks * 32 + half * 8;
                const unsigned short* hp1 = hp0 + 16 * HPAD;
                ha0.q[0] = *(const uint4*)(hp0);
                ha0.q[1] = *(const uint4*)(hp0 + 16);
                ha1.q[0] = *(const uint4*)(hp1);
                ha1.q[1] = *(const uint4*)(hp1 + 16);
            }
            const int kb2 = mc * 4 + ks;
            BFrag wb[2];
            if (PACKED) {
                const unsigned int* pb2 = pw2e + ((size_t)(wave * 6) * KB2 + kb2) * 256 + lane * 8;
                wb[0].q[0] = *(const uint4*)(pb2);
                wb[0].q[1] = *(const uint4*)(pb2 + 4);
            } else {
#pragma unroll
                for (int v = 0; v < 8; ++v) {
                    int kk = kb2 * 32 + half * 8 + (v >> 2) * 16 + (v & 3) * 2;
                    const float* p = W2e + (size_t)kk * DDIM + wave * 96 + n16;
                    wb[0].u[v] = pack2(p[0], p[DDIM]);
                }
            }
#pragma unroll
            for (int j = 0; j < 6; ++j) {
                const int cur = j & 1, nxt = cur ^ 1;
                if (j + 1 < 6) {   // double-buffer next col tile's B
                    if (PACKED) {
                        const unsigned int* pb2 =
                            pw2e + ((size_t)(wave * 6 + j + 1) * KB2 + kb2) * 256 + lane * 8;
                        wb[nxt].q[0] = *(const uint4*)(pb2);
                        wb[nxt].q[1] = *(const uint4*)(pb2 + 4);
                    } else {
                        const int dcol = (wave * 6 + j + 1) * 16 + n16;
#pragma unroll
                        for (int v = 0; v < 8; ++v) {
                            int kk = kb2 * 32 + half * 8 + (v >> 2) * 16 + (v & 3) * 2;
                            const float* p = W2e + (size_t)kk * DDIM + dcol;
                            wb[nxt].u[v] = pack2(p[0], p[DDIM]);
                        }
                    }
                }
                acc[0][j] = __builtin_amdgcn_wmma_f32_16x16x32_bf16(
                    false, ha0.v, false, wb[cur].v, (short)0, acc[0][j], false, false);
                acc[1][j] = __builtin_amdgcn_wmma_f32_16x16x32_bf16(
                    false, ha1.v, false, wb[cur].v, (short)0, acc[1][j], false, false);
            }
        }
        __syncthreads();
    }

    // ---------------- epilogue: +b2, combine-weight, atomic accumulate into y --
    for (int rt = 0; rt < 2; ++rt) {
#pragma unroll
        for (int j = 0; j < 6; ++j) {
            const int dcol = (wave * 6 + j) * 16 + n16;
            const float bias = b2e[dcol];
#pragma unroll
            for (int v = 0; v < 8; ++v) {
                int rl = rt * 16 + half * 8 + v;
                float val = (acc[rt][j][v] + bias) * cw_s[rl];
                atomicAdd(y + (size_t)(row0 + rl) * DDIM + dcol, val);
            }
        }
    }
}

// ---------------------------------------------------------------- host entry
extern "C" void kernel_launch(void* const* d_in, const int* in_sizes, int n_in,
                              void* d_out, int out_size, void* d_ws, size_t ws_size,
                              hipStream_t stream) {
    const float* x  = (const float*)d_in[0];
    const float* Wg = (const float*)d_in[1];
    const float* bg = (const float*)d_in[2];
    const float* We = (const float*)d_in[3];
    const float* be = (const float*)d_in[4];
    const float* W1 = (const float*)d_in[5];
    const float* b1 = (const float*)d_in[6];
    const float* W2 = (const float*)d_in[7];
    const float* b2 = (const float*)d_in[8];
    float* y = (float*)d_out;

    size_t off = 0;
    unsigned short* xbf = (unsigned short*)d_ws;                 off += (size_t)NTOK * DDIM * 2;
    float* combine = (float*)((char*)d_ws + off);                off += (size_t)NTOK * GE * 4;
    const size_t pw_elems = (size_t)GE * DDIM * MDIM / 2;        // dwords per packed weight
    unsigned int* pw1 = (unsigned int*)((char*)d_ws + off);
    unsigned int* pw2 = pw1 + pw_elems;
    const bool packed = ws_size >= off + 2 * pw_elems * sizeof(unsigned int);

    {
        int n = NTOK * DDIM;
        zero_kernel<<<(n + 255) / 256, 256, 0, stream>>>(y, n);
        cvt_kernel<<<(n + 255) / 256, 256, 0, stream>>>(x, xbf, n);
    }
    gate_kernel<<<NTOK, 128, 0, stream>>>(x, Wg, bg, We, be, combine);

    dim3 grid(NTOK / TILE_N, GE);
    if (packed) {
        int n1 = GE * NB1 * KB1 * 256;   // == pw_elems
        int n2 = GE * NB2 * KB2 * 256;
        pack_w1_kernel<<<n1 / 256, 256, 0, stream>>>(W1, pw1);
        pack_w2_kernel<<<n2 / 256, 256, 0, stream>>>(W2, pw2);
        moe_ffn_kernel<true><<<grid, 256, 0, stream>>>(xbf, W1, b1, W2, b2, pw1, pw2, combine, y);
    } else {
        moe_ffn_kernel<false><<<grid, 256, 0, stream>>>(xbf, W1, b1, W2, b2, pw1, pw2, combine, y);
    }
}
